// I2DAttention_20332375179265
// MI455X (gfx1250) — matrix operand order repaired and checked by reading, not compiled, not run on hardware
//
#include <hip/hip_runtime.h>

typedef _Float16 v16h __attribute__((ext_vector_type(16)));
typedef _Float16 v8h  __attribute__((ext_vector_type(8)));
typedef float    v8f  __attribute__((ext_vector_type(8)));

#define NEGINF (-3.402823466e38f)

static __device__ __forceinline__ v8f wmma_f16(v16h a, v16h b, v8f c) {
  // D = A(16x32 f16) x B(32x16 f16) + C(16x16 f32)
  return __builtin_amdgcn_wmma_f32_16x16x32_f16(false, a, false, b, (short)0, c,
                                                false, false);
}

// ---------------------------------------------------------------------------
// Kernel 1: per-head 64x64 projection.  rows = flattened [batch*tokens*H],
// dst[row][n] = sum_d src[row][d] * W[n][d]   (y = x @ W^T), f16 output.
// One wave per 16-row tile; 4 N-tiles x 2 K-steps = 8 WMMAs per wave.
// ---------------------------------------------------------------------------
__global__ __launch_bounds__(32)
void proj64_kernel(const float* __restrict__ src, const float* __restrict__ W,
                   _Float16* __restrict__ dst) {
  const int tile  = blockIdx.x;
  const int lane  = threadIdx.x;
  const int lhalf = lane & 15;
  const int lhi   = lane >> 4;
  const int row   = tile * 16 + lhalf;  // A-matrix row (M)

  // A fragments for both K-steps (contiguous K layout: k = 16*lhi + j)
  v16h A[2];
#pragma unroll
  for (int kk = 0; kk < 2; ++kk) {
    const float* p0 = src + (size_t)row * 64 + kk * 32 + lhi * 16;
    const float* p1 = p0 + 8;
#pragma unroll
    for (int j = 0; j < 8; ++j) {
      A[kk][j]     = (_Float16)p0[j];
      A[kk][j + 8] = (_Float16)p1[j];
    }
  }

#pragma unroll
  for (int nt = 0; nt < 4; ++nt) {
    const int n = nt * 16 + lhalf;      // B-matrix column (N)
    v8f acc = {};
#pragma unroll
    for (int kk = 0; kk < 2; ++kk) {
      const float* bp = W + (size_t)n * 64 + kk * 32 + lhi * 16;
      v16h B;
#pragma unroll
      for (int j = 0; j < 16; ++j) B[j] = (_Float16)bp[j];
      acc = wmma_f16(A[kk], B, acc);
    }
#pragma unroll
    for (int r = 0; r < 8; ++r) {
      const int M = (lane < 16) ? r : (r + 8);
      dst[(size_t)(tile * 16 + M) * 64 + nt * 16 + lhalf] = (_Float16)acc[r];
    }
  }
}

// ---------------------------------------------------------------------------
// Kernel 2: per-(i,t,h) 197x30 energy via WMMA, dual softmax, attention-map
// streaming writes, and fused pooled reductions.
// ---------------------------------------------------------------------------
__global__ __launch_bounds__(256)
void attn_kernel(const _Float16* __restrict__ qf, const _Float16* __restrict__ kf,
                 const _Float16* __restrict__ vf, const _Float16* __restrict__ tvf,
                 const unsigned char* __restrict__ tmask,
                 float* __restrict__ att, float* __restrict__ tatt,
                 _Float16* __restrict__ sout, _Float16* __restrict__ stext) {
  constexpr int Q = 197, K = 30, H = 8, T = 64, D = 64;
  constexpr int QP = 208, KP = 32;
  __shared__ float Esc[QP * KP];
  __shared__ float rowMax[QP], rowDen[QP], colMax[KP], colDen[KP];
  __shared__ float rsum[QP], csum[KP];

  const int bid = blockIdx.x;
  const int h = bid % H;
  const int t = (bid / H) % T;
  const int i = bid / (H * T);

  const int tid   = threadIdx.x;
  const int wave  = tid >> 5;
  const int lane  = tid & 31;
  const int lhalf = lane & 15;
  const int lhi   = lane >> 4;

  // ---- Phase 1: energy tiles (13 Q-tiles x 2 K-tiles), masked+scaled to LDS
  for (int qt = wave; qt < 13; qt += 8) {
    const int qrow = qt * 16 + lhalf;
    const int qrc  = (qrow < Q) ? qrow : (Q - 1);
    const _Float16* abase = qf + ((size_t)(i * Q + qrc) * H + h) * D;
    v16h A[2];
#pragma unroll
    for (int kk = 0; kk < 2; ++kk) {
      A[kk] = *(const v16h*)(abase + kk * 32 + lhi * 16);
    }
#pragma unroll
    for (int nt = 0; nt < 2; ++nt) {
      const int kcol = nt * 16 + lhalf;
      const int kcc  = (kcol < K) ? kcol : (K - 1);
      v8f acc = {};
#pragma unroll
      for (int kk = 0; kk < 2; ++kk) {
        const v16h* bp = (const v16h*)(kf + ((size_t)(t * K + kcc) * H + h) * D +
                                       kk * 32 + lhi * 16);
        acc = wmma_f16(A[kk], *bp, acc);
      }
      // cls column always unmasked; pad columns / pad rows -> NEG
      const bool mok =
          (kcol == 0) || (kcol < K && tmask[t * (K - 1) + (kcol - 1)] != 0);
#pragma unroll
      for (int r = 0; r < 8; ++r) {
        const int M  = (lane < 16) ? r : (r + 8);
        const int qr = qt * 16 + M;
        const float val =
            ((qr < Q) && mok) ? acc[r] * 0.04419417382415922f : NEGINF;
        Esc[qr * KP + (nt * 16 + lhalf)] = val;
      }
    }
  }
  __syncthreads();

  // ---- Phase 2: softmax stats (rows: over k ; cols: over q)
  if (tid < QP) {
    float m = NEGINF;
    for (int c0 = 0; c0 < KP; ++c0) m = fmaxf(m, Esc[tid * KP + c0]);
    float d = 0.f;
    for (int c0 = 0; c0 < KP; ++c0) d += __expf(Esc[tid * KP + c0] - m);
    rowMax[tid] = m; rowDen[tid] = d; rsum[tid] = 0.f;
  }
  if (tid < KP) {
    float m = NEGINF;
    for (int r = 0; r < QP; ++r) m = fmaxf(m, Esc[r * KP + tid]);
    float d = 0.f;
    for (int r = 0; r < QP; ++r) d += __expf(Esc[r * KP + tid] - m);
    colMax[tid] = m; colDen[tid] = d; csum[tid] = 0.f;
  }
  __syncthreads();

  // ---- Phase 3: coalesced NT writes of both maps + LDS-atomic reductions
  const size_t base = (((size_t)(i * T + t) * H + h) * Q) * K;
  for (int idx = tid; idx < Q * K; idx += 256) {
    const int r = idx / K, c0 = idx % K;
    const float x  = Esc[r * KP + c0];
    const float a  = __expf(x - rowMax[r]) / rowDen[r];
    const float ta = __expf(x - colMax[c0]) / colDen[c0];
    __builtin_nontemporal_store(a, &att[base + idx]);
    __builtin_nontemporal_store(ta, &tatt[base + idx]);
    atomicAdd(&csum[c0], a);   // colsum of attention (for i2t pooled branch)
    atomicAdd(&rsum[r], ta);   // rowsum of text_attention (for t2i branch)
  }
  __syncthreads();

  // ---- Phase 4: fused pooled contractions (mean commuted past W_out)
  if (tid < D) {
    float a0 = 0.f;
    for (int c0 = 0; c0 < K; ++c0)
      a0 += csum[c0] * (float)vf[((size_t)(t * K + c0) * H + h) * D + tid];
    sout[((size_t)(i * T + t) * H + h) * D + tid] = (_Float16)a0;

    float a1 = 0.f;
    for (int r = 0; r < Q; ++r)
      a1 += rsum[r] * (float)tvf[((size_t)(i * Q + r) * H + h) * D + tid];
    stext[((size_t)(i * T + t) * H + h) * D + tid] = (_Float16)a1;
  }
}

// ---------------------------------------------------------------------------
// Kernel 3: out[i,t,:] = scale * s[i,t,:] @ W^T + b   (two 1024x512x512 GEMMs)
// One wave per 16x16 output tile; K-loop of 16 WMMAs.
// ---------------------------------------------------------------------------
__global__ __launch_bounds__(32)
void final_gemm_kernel(const _Float16* __restrict__ sout,
                       const _Float16* __restrict__ stext,
                       const float* __restrict__ Wo, const float* __restrict__ Wot,
                       const float* __restrict__ bo, const float* __restrict__ bot,
                       float* __restrict__ out0, float* __restrict__ out1) {
  const int mt = blockIdx.x, nt = blockIdx.y, which = blockIdx.z;
  const _Float16* A = which ? stext : sout;
  const float* W    = which ? Wot : Wo;
  const float* b    = which ? bot : bo;
  float* dst        = which ? out1 : out0;
  const float scale = which ? (1.0f / 30.0f) : (1.0f / 197.0f);

  const int lane = threadIdx.x, lhalf = lane & 15, lhi = lane >> 4;
  const int m = mt * 16 + lhalf;
  const int n = nt * 16 + lhalf;

  v8f acc = {};
  for (int kk = 0; kk < 16; ++kk) {
    v16h Af = *(const v16h*)(A + (size_t)m * 512 + kk * 32 + lhi * 16);
    const float* bp = W + (size_t)n * 512 + kk * 32 + lhi * 16;
    v16h Bf;
#pragma unroll
    for (int j = 0; j < 16; ++j) Bf[j] = (_Float16)bp[j];
    acc = wmma_f16(Af, Bf, acc);
  }
#pragma unroll
  for (int r = 0; r < 8; ++r) {
    const int M  = (lane < 16) ? r : (r + 8);
    const int gm = mt * 16 + M, gn = nt * 16 + lhalf;
    dst[(size_t)gm * 512 + gn] = acc[r] * scale + b[gn];
  }
}

// ---------------------------------------------------------------------------
extern "C" void kernel_launch(void* const* d_in, const int* in_sizes, int n_in,
                              void* d_out, int out_size, void* d_ws,
                              size_t ws_size, hipStream_t stream) {
  (void)in_sizes; (void)n_in; (void)out_size; (void)ws_size;

  const float* image = (const float*)d_in[0];   // [16,197,512]
  const float* text  = (const float*)d_in[1];   // [64,30,512]
  const unsigned char* tmask = (const unsigned char*)d_in[2];  // [64,29] bool
  const float* Wq  = (const float*)d_in[3];
  const float* Wk  = (const float*)d_in[4];
  const float* Wv  = (const float*)d_in[5];
  const float* Wo  = (const float*)d_in[6];
  const float* bo  = (const float*)d_in[7];
  const float* Wtv = (const float*)d_in[8];
  const float* Wot = (const float*)d_in[9];
  const float* bot = (const float*)d_in[10];

  // workspace carve-out (~12.5 MB)
  char* ws = (char*)d_ws;
  size_t off = 0;
  auto alloc = [&](size_t bytes) {
    char* p = ws + off;
    off += (bytes + 255) & ~(size_t)255;
    return p;
  };
  _Float16* qf     = (_Float16*)alloc(25216ull * 64 * 2);  // [I*Q*H, D]
  _Float16* tvf    = (_Float16*)alloc(25216ull * 64 * 2);
  _Float16* kf     = (_Float16*)alloc(15360ull * 64 * 2);  // [T*K*H, D]
  _Float16* vf     = (_Float16*)alloc(15360ull * 64 * 2);
  _Float16* soutf  = (_Float16*)alloc(1024ull * 512 * 2);  // [I*T, E]
  _Float16* stextf = (_Float16*)alloc(1024ull * 512 * 2);

  // output tuple layout (flat concat, f32)
  float* out0 = (float*)d_out;                // [16,64,512]
  float* out1 = out0 + 16 * 64 * 512;         // [16,64,512]
  float* att  = out1 + 16 * 64 * 512;         // [16,64,8,197,30]
  float* tatt = att + 48414720ull;            // [16,64,8,197,30]

  // 1) per-head projections (f32 -> f16 operands), WMMA
  proj64_kernel<<<1576, 32, 0, stream>>>(image, Wq, qf);
  proj64_kernel<<<1576, 32, 0, stream>>>(image, Wtv, tvf);
  proj64_kernel<<<960, 32, 0, stream>>>(text, Wk, kf);
  proj64_kernel<<<960, 32, 0, stream>>>(text, Wv, vf);

  // 2) energy + dual softmax + streaming attention writes + fused pooling
  attn_kernel<<<16 * 64 * 8, 256, 0, stream>>>(qf, kf, vf, tvf, tmask, att,
                                               tatt, soutf, stextf);

  // 3) pooled output projections with scale+bias epilogue
  final_gemm_kernel<<<dim3(64, 32, 2), 32, 0, stream>>>(soutf, stextf, Wo, Wot,
                                                        bo, bot, out0, out1);
}
